// MHLA_Normed_Torch_83803401880229
// MI455X (gfx1250) — matrix operand order repaired and checked by reading, not compiled, not run on hardware
//
#include <hip/hip_runtime.h>

// MI455X / gfx1250: windowed multi-head linear attention.
// All matmuls on v_wmma_f32_16x16x32_f16. All WMMA operands are stored in
// fragment-major layouts so every fragment = two 16B vector loads (b128).
// Weights swizzled once (L2-resident); out-proj GEMM stages its A tile with
// the Tensor Data Mover (tensor_load_to_lds + s_wait_tensorcnt), double-buffered.

typedef __attribute__((ext_vector_type(16))) _Float16 v16h;
typedef __attribute__((ext_vector_type(8)))  _Float16 v8h;
typedef __attribute__((ext_vector_type(8)))  float    v8f;

#define EPS_ATTN 1e-6f

// ---------------- TDM availability (probe-verified builtin, arity differs) ---------------
#if defined(__has_builtin)
#if __has_builtin(__builtin_amdgcn_tensor_load_to_lds) && __has_builtin(__builtin_amdgcn_s_wait_tensorcnt)
#define HAVE_TDM 1
#endif
#endif
#ifndef HAVE_TDM
#define HAVE_TDM 0
#endif

#if HAVE_TDM
typedef unsigned int u32x4 __attribute__((ext_vector_type(4)));
typedef int          i32x4 __attribute__((ext_vector_type(4)));
typedef int          i32x8 __attribute__((ext_vector_type(8)));

// 2D tile load: d0 elements/row (2B each), d1 rows, row stride `stride` elements.
// D# built per CDNA5 ISA 8.3/8.4: group0 = {flags, lds_addr, global_addr, type=2},
// group1 = {data_size=2B, tensor_dim0/1, tile_dim0/1, dim0_stride}.
__device__ __forceinline__ void tdm_load_tile_2d(unsigned ldsAddr, const void* g,
                                                 unsigned d0, unsigned d1, unsigned stride) {
  unsigned long long ga = (unsigned long long)g;
  u32x4 g0;
  g0[0] = 1u;                                                   // count=1, user mode
  g0[1] = ldsAddr;                                              // LDS byte address
  g0[2] = (unsigned)(ga & 0xffffffffu);                         // global_addr[31:0]
  g0[3] = (unsigned)((ga >> 32) & 0x01ffffffu) | 0x80000000u;   // addr[56:32] | type=2
  i32x8 g1;
  g1[0] = (int)(1u << 16);                                      // data_size=1 (2 bytes)
  g1[1] = (int)((d0 & 0xffffu) << 16);                          // tensor_dim0 lo16
  g1[2] = (int)((d0 >> 16) | ((d1 & 0xffffu) << 16));           // dim0 hi | dim1 lo
  g1[3] = (int)(((d1 >> 16) & 0xffffu) | ((d0 & 0xffffu) << 16)); // dim1 hi | tile_dim0
  g1[4] = (int)(d1 & 0xffffu);                                  // tile_dim1 (tile_dim2=0)
  g1[5] = (int)stride;                                          // dim0_stride lo32
  g1[6] = 0;
  g1[7] = 0;
  i32x4 z4 = {0, 0, 0, 0};
#if __clang_major__ >= 23
  i32x8 z8 = {0, 0, 0, 0, 0, 0, 0, 0};
  __builtin_amdgcn_tensor_load_to_lds(g0, g1, z4, z4, z8, 0);
#else
  __builtin_amdgcn_tensor_load_to_lds(g0, g1, z4, z4, 0);
#endif
}
#endif  // HAVE_TDM

// ---------------- fragment helpers (vectorized; contiguous k = 16*(lane/16) + s in A and B) ----------------
__device__ __forceinline__ v16h combine16(v8h lo, v8h hi) {
  v16h f;
#pragma unroll
  for (int j = 0; j < 8; ++j) { f[j] = lo[j]; f[j + 8] = hi[j]; }
  return f;
}

__device__ __forceinline__ v8f wmma_fma(v16h a, v16h b, v8f c) {
  return __builtin_amdgcn_wmma_f32_16x16x32_f16(false, a, false, b, (short)0, c,
                                                false, false);
}

__device__ __forceinline__ v8f zero_v8f() {
  v8f z;
#pragma unroll
  for (int j = 0; j < 8; ++j) z[j] = 0.0f;
  return z;
}

// A fragment from row-major A[m][k], ld multiple of 8. Two b128 loads.
__device__ __forceinline__ v16h frag_a_rm(const _Float16* p, int ld, int mr, int kb, int lane) {
  const _Float16* q = p + (mr + (lane & 15)) * ld + kb + ((lane >> 4) << 4);
  return combine16(*(const v8h*)q, *(const v8h*)(q + 8));
}

// B fragment from n-major storage S[n][k] (B[k][n] = S[n*ld + k]). Two b128 loads.
__device__ __forceinline__ v16h frag_b_nm(const _Float16* p, int ld, int kb, int nc, int lane) {
  const _Float16* q = p + (nc + (lane & 15)) * ld + kb + ((lane >> 4) << 4);
  return combine16(*(const v8h*)q, *(const v8h*)(q + 8));
}

// B fragment from a pre-swizzled 512-half fragment panel.
__device__ __forceinline__ v16h frag_b_panel(const _Float16* p, int lane) {
  const _Float16* q = p + (lane & 15) * 32 + ((lane >> 4) << 4);
  return combine16(*(const v8h*)q, *(const v8h*)(q + 8));
}

// ---------------- weight swizzles (run once per launch; L2-resident results) ----------
// B panels: dst[((kc*Nt+nt)*16+col)*32 + kk] = src[(kc*32+kk)*N + nt*16+col]
__global__ __launch_bounds__(256) void conv_b_sw_kn(const float* __restrict__ src,
                                                    _Float16* __restrict__ dst,
                                                    int N, int Nt, int total) {
  int d = blockIdx.x * 256 + threadIdx.x;
  if (d >= total) return;
  int kk  = d & 31;
  int col = (d >> 5) & 15;
  int nt  = (d >> 9) % Nt;
  int kc  = d / (512 * Nt);
  dst[d] = (_Float16)src[(kc * 32 + kk) * N + nt * 16 + col];
}

// A panels for piece_w (64x64): dst[((mt*2+kc)*16+row)*32+kk] = src[(mt*16+row)*64 + kc*32+kk]
__global__ __launch_bounds__(256) void conv_a_sw_kn(const float* __restrict__ src,
                                                    _Float16* __restrict__ dst) {
  int d = blockIdx.x * 256 + threadIdx.x;  // 4096
  int kk = d & 31, row = (d >> 5) & 15, kc = (d >> 9) & 1, mt = d >> 10;
  dst[d] = (_Float16)src[(mt * 16 + row) * 64 + kc * 32 + kk];
}

// ---------------- LayerNorm row stats (wave32 per row) ----------------
__global__ __launch_bounds__(256) void ln_stats_kn(const float* __restrict__ x,
                                                   float* __restrict__ mu,
                                                   float* __restrict__ rstd) {
  int row  = blockIdx.x * 8 + (threadIdx.x >> 5);
  int lane = threadIdx.x & 31;
  const float* p = x + (long)row * 256;
  float s = 0.f, s2 = 0.f;
#pragma unroll
  for (int j = 0; j < 8; ++j) {
    float v = p[j * 32 + lane];
    s += v; s2 += v * v;
  }
#pragma unroll
  for (int o = 16; o > 0; o >>= 1) {
    s  += __shfl_xor(s, o, 32);
    s2 += __shfl_xor(s2, o, 32);
  }
  if (lane == 0) {
    float m = s * (1.f / 256.f);
    mu[row]   = m;
    rstd[row] = rsqrtf(s2 * (1.f / 256.f) - m * m + 1e-5f);
  }
}

// ---------------- LN-fused QKV GEMM (M=100352, K=256, N=768) ----------------
// A staged in LDS (LN applied on the fly); B fragments straight from swizzled global.
// Epilogue routes: q -> row-major (relu+eps), k/v -> transposed [bn*256+c][w] (w padded 64).
__global__ __launch_bounds__(256) void ln_qkv_gemm_kn(
    const float* __restrict__ x, const float* __restrict__ mu, const float* __restrict__ rstd,
    const float* __restrict__ g, const float* __restrict__ bb,
    const _Float16* __restrict__ wq_sw,
    _Float16* __restrict__ qh, _Float16* __restrict__ k_sw, _Float16* __restrict__ v_sw) {
  __shared__ alignas(32) _Float16 sA[128 * 32];
  int tid = threadIdx.x, lane = tid & 31, wv = tid >> 5;
  long m0 = (long)blockIdx.x * 128;
  int  by = blockIdx.y;

  v8f acc[4];
#pragma unroll
  for (int t = 0; t < 4; ++t) acc[t] = zero_v8f();

  for (int ch = 0; ch < 8; ++ch) {
    int k0 = ch * 32;
#pragma unroll
    for (int i = 0; i < 16; ++i) {
      int idx = i * 256 + tid;
      int row = idx >> 5, kk = idx & 31;
      long gm = m0 + row;
      float xv = x[gm * 256 + k0 + kk];
      sA[idx] = (_Float16)((xv - mu[gm]) * rstd[gm] * g[k0 + kk] + bb[k0 + kk]);
    }
    if (ch + 1 < 8) __builtin_prefetch(&wq_sw[((ch + 1) * 48 + by * 4) * 512], 0, 0);
    __syncthreads();
    v16h a = frag_a_rm(sA, 32, wv * 16, 0, lane);
#pragma unroll
    for (int t = 0; t < 4; ++t) {
      v16h bf = frag_b_panel(wq_sw + (ch * 48 + by * 4 + t) * 512, lane);
      acc[t] = wmma_fma(a, bf, acc[t]);
    }
    __syncthreads();
  }

  int hi8 = (lane >> 4) << 3, cl = lane & 15;
#pragma unroll
  for (int t = 0; t < 4; ++t) {
#pragma unroll
    for (int r = 0; r < 8; ++r) {
      long gm = m0 + wv * 16 + r + hi8;
      int  gc = by * 64 + t * 16 + cl;
      float v = acc[t][r];
      if (gc < 256) {
        qh[gm * 256 + gc] = (_Float16)(fmaxf(v, 0.f) + EPS_ATTN);
      } else {
        int bn = (int)(gm / 49), w = (int)(gm - (long)bn * 49);
        if (gc < 512)
          k_sw[((long)bn * 256 + (gc - 256)) * 64 + w] = (_Float16)(fmaxf(v, 0.f) + EPS_ATTN);
        else
          v_sw[((long)bn * 256 + (gc - 512)) * 64 + w] = (_Float16)v;
      }
    }
  }
}

// ---------------- per-window kv = k^T v, ksum, z = q.ksum ----------------
// Wave per (b,n,h); no LDS, no barriers; all fragments are contiguous b128 global loads.
__global__ __launch_bounds__(256) void window_kv_kn(
    const _Float16* __restrict__ k_sw, const _Float16* __restrict__ v_sw,
    const _Float16* __restrict__ qh,
    _Float16* __restrict__ kv_sw, float* __restrict__ z) {
  int lane = threadIdx.x & 31;
  int gid = blockIdx.x * 8 + (threadIdx.x >> 5);
  int h = gid & 7, bn = gid >> 3;
  const _Float16* kp = k_sw + ((long)bn * 256 + h * 32) * 64;  // A[d][w], row-major ld=64
  const _Float16* vp = v_sw + ((long)bn * 256 + h * 32) * 64;  // B[w][e] = vp[e*64+w]

  v8f acc[2][2];
#pragma unroll
  for (int i = 0; i < 2; ++i)
#pragma unroll
    for (int j = 0; j < 2; ++j) acc[i][j] = zero_v8f();

#pragma unroll
  for (int kc = 0; kc < 2; ++kc) {
    int kb = kc * 32;
    v16h a0 = frag_a_rm(kp, 64, 0,  kb, lane);
    v16h a1 = frag_a_rm(kp, 64, 16, kb, lane);
    v16h b0 = frag_b_nm(vp, 64, kb, 0,  lane);
    v16h b1 = frag_b_nm(vp, 64, kb, 16, lane);
    acc[0][0] = wmma_fma(a0, b0, acc[0][0]);
    acc[0][1] = wmma_fma(a0, b1, acc[0][1]);
    acc[1][0] = wmma_fma(a1, b0, acc[1][0]);
    acc[1][1] = wmma_fma(a1, b1, acc[1][1]);
  }

  // ksum[d] for d = lane
  float ks = 0.f;
  {
    const _Float16* kr = kp + lane * 64;
    for (int w = 0; w < 49; ++w) ks += (float)kr[w];
  }
  // z[w,h] = sum_d q[w,h,d] * ksum[d]  (cross-lane reduce)
  long qbase = (long)bn * 49 * 256 + h * 32;
  for (int w = 0; w < 49; ++w) {
    float p = (float)qh[qbase + (long)w * 256 + lane] * ks;
#pragma unroll
    for (int o = 16; o > 0; o >>= 1) p += __shfl_xor(p, o, 32);
    if (lane == 0) z[((long)bn * 49 + w) * 8 + h] = p;
  }

  // store kv n-major for the piece-mix B fragments: kv_sw[(bh*1024 + d*32+e)*64 + n]
  int hi8 = (lane >> 4) << 3, cl = lane & 15;
  int b = bn >> 6, n = bn & 63;
  long bh = (long)b * 8 + h;
#pragma unroll
  for (int mi = 0; mi < 2; ++mi)
#pragma unroll
    for (int ni = 0; ni < 2; ++ni)
#pragma unroll
      for (int r = 0; r < 8; ++r) {
        int d = mi * 16 + r + hi8, e = ni * 16 + cl;
        kv_sw[(bh * 1024 + d * 32 + e) * 64 + n] = (_Float16)acc[mi][ni][r];
      }
}

// ---------------- kv' = P @ kv (GEMM M=64,K=64 over N=1024 cols per (b,h)) ----------------
// Wave per (bh, 16-col tile); A panels + B (n-major) straight from global.
__global__ __launch_bounds__(256) void piece_mix_kv_kn(
    const _Float16* __restrict__ pA, const _Float16* __restrict__ kv_sw,
    _Float16* __restrict__ kvp_sw) {
  int lane = threadIdx.x & 31;
  int gid = blockIdx.x * 8 + (threadIdx.x >> 5);
  long bh = gid >> 6;
  int cb = (gid & 63) * 16;
  const _Float16* bp = kv_sw + bh * 65536;

  v8f acc[4];
#pragma unroll
  for (int t = 0; t < 4; ++t) acc[t] = zero_v8f();

#pragma unroll
  for (int kc = 0; kc < 2; ++kc) {
    v16h bf = frag_b_nm(bp, 64, kc * 32, cb, lane);
#pragma unroll
    for (int t = 0; t < 4; ++t) {
      v16h a = frag_a_rm(pA + (t * 2 + kc) * 512, 32, 0, 0, lane);
      acc[t] = wmma_fma(a, bf, acc[t]);
    }
  }

  int hi8 = (lane >> 4) << 3, cl = lane & 15;
  int col = cb + cl, d = col >> 5, e = col & 31;
#pragma unroll
  for (int t = 0; t < 4; ++t)
#pragma unroll
    for (int r = 0; r < 8; ++r) {
      int m = t * 16 + r + hi8;
      // d-major for the attention B fragments: kvp_sw[(bh*64+m)*1024 + e*32 + d]
      kvp_sw[(bh * 64 + m) * 1024 + e * 32 + d] = (_Float16)acc[t][r];
    }
}

// ---------------- z' = eps + P @ z (small, VALU) ----------------
__global__ __launch_bounds__(256) void piece_mix_z_kn(const float* __restrict__ P,
                                                      const float* __restrict__ z,
                                                      float* __restrict__ zp) {
  __shared__ float sP[4096];
  for (int idx = threadIdx.x; idx < 4096; idx += 256) sP[idx] = P[idx];
  __syncthreads();
  long idx = (long)blockIdx.x * 256 + threadIdx.x;   // 802816 total
  int h = idx & 7;
  long t2 = idx >> 3;
  int w = (int)(t2 % 49);
  long t3 = t2 / 49;
  int m = (int)(t3 & 63), b = (int)(t3 >> 6);
  long zbase = (long)b * 25088 + (long)w * 8 + h;
  float s = EPS_ATTN;
  for (int n = 0; n < 64; ++n) s += sP[m * 64 + n] * z[zbase + (long)n * 392];
  zp[idx] = s;
}

// ---------------- out = (q @ kv') / z' ----------------
// Wave per (b,m,h); no LDS; q rows row-major (16B-aligned runs), kv' d-major.
__global__ __launch_bounds__(256) void attn_apply_kn(
    const _Float16* __restrict__ qh, const _Float16* __restrict__ kvp_sw,
    const float* __restrict__ zp, _Float16* __restrict__ attn) {
  int lane = threadIdx.x & 31;
  int gid = blockIdx.x * 8 + (threadIdx.x >> 5);
  int h = gid & 7, m = (gid >> 3) & 63, b = gid >> 9;
  long qbase = (long)(b * 64 + m) * 49 * 256;
  const _Float16* qp = qh + qbase + h * 32;
  const _Float16* bp = kvp_sw + ((long)(b * 8 + h) * 64 + m) * 1024;

  v8f acc[4][2];
#pragma unroll
  for (int t = 0; t < 4; ++t)
#pragma unroll
    for (int ni = 0; ni < 2; ++ni) acc[t][ni] = zero_v8f();

#pragma unroll
  for (int t = 0; t < 4; ++t) {
    v16h a = frag_a_rm(qp, 256, t * 16, 0, lane);   // rows >= 49 read slack, never stored
#pragma unroll
    for (int ni = 0; ni < 2; ++ni) {
      v16h bf = frag_b_nm(bp, 32, 0, ni * 16, lane);
      acc[t][ni] = wmma_fma(a, bf, acc[t][ni]);
    }
  }

  int hi8 = (lane >> 4) << 3, cl = lane & 15;
#pragma unroll
  for (int t = 0; t < 4; ++t)
#pragma unroll
    for (int ni = 0; ni < 2; ++ni)
#pragma unroll
      for (int r = 0; r < 8; ++r) {
        int w = t * 16 + r + hi8;
        if (w < 49) {
          int e = ni * 16 + cl;
          float zv = zp[((long)(b * 64 + m) * 49 + w) * 8 + h];
          attn[qbase + (long)w * 256 + h * 32 + e] = (_Float16)(acc[t][ni][r] / zv);
        }
      }
}

// ---------------- LePE depthwise 5x5 (reads transposed v) + add -> y (f16) ----------------
__global__ __launch_bounds__(256) void lepe_add_kn(
    const _Float16* __restrict__ v_sw, const _Float16* __restrict__ attn,
    const float* __restrict__ lw, const float* __restrict__ lb,
    _Float16* __restrict__ y) {
  long idx = (long)blockIdx.x * 256 + threadIdx.x;   // (b,n,t,c) flat
  int c = idx & 255;
  long t1 = idx >> 8;
  int t = (int)(t1 % 49);
  long t2 = t1 / 49;
  int n = (int)(t2 & 63), b = (int)(t2 >> 6);
  int hb = n >> 3, wb = n & 7;
  int p1 = t / 7, p2 = t % 7;
  int r = hb * 7 + p1, cc = wb * 7 + p2;
  float s = 0.f;
#pragma unroll
  for (int dy = 0; dy < 5; ++dy) {
    int rr = r + dy - 2;
    if (rr < 0 || rr >= 56) continue;
    int n_r = (rr / 7) * 8;
    int t_r = (rr % 7) * 7;
#pragma unroll
    for (int dx = 0; dx < 5; ++dx) {
      int c2 = cc + dx - 2;
      if (c2 < 0 || c2 >= 56) continue;
      int n2 = n_r + c2 / 7;
      int t2i = t_r + c2 % 7;
      float vv = (float)v_sw[(((long)(b * 64 + n2)) * 256 + c) * 64 + t2i];
      s += lw[c * 25 + dy * 5 + dx] * vv;
    }
  }
  s += lb[c] + (float)attn[idx];
  y[idx] = (_Float16)s;
}

// ---------------- out-proj GEMM (M=100352, K=256, N=256) + bias ----------------
// A tile (128x32, stride 256) staged via TDM (double buffered); B panels from global.
__global__ __launch_bounds__(256) void out_gemm_kn(
    const _Float16* __restrict__ y, const _Float16* __restrict__ wo_sw,
    const float* __restrict__ bo, float* __restrict__ out) {
  __shared__ alignas(32) _Float16 sA[2][128 * 32];
  int tid = threadIdx.x, lane = tid & 31, wv = tid >> 5;
  long m0 = (long)blockIdx.x * 128;
  int nb = blockIdx.y;

  v8f acc[4];
#pragma unroll
  for (int t = 0; t < 4; ++t) acc[t] = zero_v8f();

#if HAVE_TDM
  if (tid < 32)
    tdm_load_tile_2d((unsigned)(unsigned long long)(void*)&sA[0][0],
                     y + m0 * 256, 32, 128, 256);
  for (int ch = 0; ch < 8; ++ch) {
    if (tid < 32) __builtin_amdgcn_s_wait_tensorcnt(0);
    __syncthreads();
    if (ch + 1 < 8 && tid < 32)
      tdm_load_tile_2d((unsigned)(unsigned long long)(void*)&sA[(ch + 1) & 1][0],
                       y + m0 * 256 + (ch + 1) * 32, 32, 128, 256);
    const _Float16* sa = &sA[ch & 1][0];
    v16h a = frag_a_rm(sa, 32, wv * 16, 0, lane);
#pragma unroll
    for (int t = 0; t < 4; ++t) {
      v16h bf = frag_b_panel(wo_sw + (ch * 16 + nb * 4 + t) * 512, lane);
      acc[t] = wmma_fma(a, bf, acc[t]);
    }
  }
#else
  for (int ch = 0; ch < 8; ++ch) {
#pragma unroll
    for (int i = 0; i < 2; ++i) {
      int u = i * 256 + tid;                 // 512 v8h units = 128x32 halves
      int row = u >> 2, k8 = u & 3;
      *(v8h*)&sA[0][row * 32 + k8 * 8] =
          *(const v8h*)&y[(m0 + row) * 256 + ch * 32 + k8 * 8];
    }
    __syncthreads();
    v16h a = frag_a_rm(&sA[0][0], 32, wv * 16, 0, lane);
#pragma unroll
    for (int t = 0; t < 4; ++t) {
      v16h bf = frag_b_panel(wo_sw + (ch * 16 + nb * 4 + t) * 512, lane);
      acc[t] = wmma_fma(a, bf, acc[t]);
    }
    __syncthreads();
  }
#endif

  int hi8 = (lane >> 4) << 3, cl = lane & 15;
#pragma unroll
  for (int t = 0; t < 4; ++t)
#pragma unroll
    for (int r = 0; r < 8; ++r) {
      long gm = m0 + wv * 16 + r + hi8;
      int  gc = nb * 64 + t * 16 + cl;
      out[gm * 256 + gc] = acc[t][r] + bo[gc];
    }
}

// ---------------------------- launch ----------------------------
extern "C" void kernel_launch(void* const* d_in, const int* in_sizes, int n_in,
                              void* d_out, int out_size, void* d_ws, size_t ws_size,
                              hipStream_t stream) {
  (void)in_sizes; (void)n_in; (void)out_size; (void)ws_size;
  const float* x    = (const float*)d_in[0];
  const float* ln_g = (const float*)d_in[1];
  const float* ln_b = (const float*)d_in[2];
  const float* wqkv = (const float*)d_in[3];
  const float* lw   = (const float*)d_in[4];
  const float* lb   = (const float*)d_in[5];
  const float* pw   = (const float*)d_in[6];
  const float* wout = (const float*)d_in[7];
  const float* bout = (const float*)d_in[8];
  float* out = (float*)d_out;

  const long M  = 32L * 64 * 49;          // 100352 token rows
  const long BN = 2048;                   // b*n windows

  char* ws = (char*)d_ws;
  size_t off = 0;
  auto alloc = [&](size_t bytes) -> void* {
    off = (off + 255) & ~(size_t)255;
    void* p = ws + off;
    off += bytes;
    return p;
  };

  float*    mu     = (float*)alloc(M * 4);
  float*    rstd   = (float*)alloc(M * 4);
  _Float16* wq_sw  = (_Float16*)alloc((size_t)196608 * 2);
  _Float16* wo_sw  = (_Float16*)alloc((size_t)65536 * 2);
  _Float16* pA     = (_Float16*)alloc((size_t)4096 * 2);
  _Float16* qh     = (_Float16*)alloc((size_t)(M * 256 + 4096) * 2);  // +slack rows
  _Float16* k_sw   = (_Float16*)alloc((size_t)BN * 256 * 64 * 2);     // [bn*256+c][w(64)]
  _Float16* v_sw   = (_Float16*)alloc((size_t)BN * 256 * 64 * 2);
  _Float16* kv_sw  = (_Float16*)alloc((size_t)256 * 1024 * 64 * 2);   // [(bh*1024+col)][n(64)]
  _Float16* kvp_sw = (_Float16*)alloc((size_t)256 * 64 * 1024 * 2);   // [(bh*64+m)][e*32+d]
  float*    zb     = (float*)alloc((size_t)M * 8 * 4);
  float*    zpb    = (float*)alloc((size_t)M * 8 * 4);
  _Float16* attn   = (_Float16*)alloc((size_t)M * 256 * 2);
  _Float16* yb     = (_Float16*)alloc((size_t)M * 256 * 2);

  // Zero k/v transposed buffers so the w=49..63 K-padding reads zeros.
  hipMemsetAsync(k_sw, 0, (size_t)BN * 256 * 64 * 2, stream);
  hipMemsetAsync(v_sw, 0, (size_t)BN * 256 * 64 * 2, stream);

  conv_b_sw_kn<<<(196608 + 255) / 256, 256, 0, stream>>>(wqkv, wq_sw, 768, 48, 196608);
  conv_b_sw_kn<<<(65536 + 255) / 256, 256, 0, stream>>>(wout, wo_sw, 256, 16, 65536);
  conv_a_sw_kn<<<16, 256, 0, stream>>>(pw, pA);

  ln_stats_kn<<<(int)(M / 8), 256, 0, stream>>>(x, mu, rstd);

  ln_qkv_gemm_kn<<<dim3(784, 12), 256, 0, stream>>>(x, mu, rstd, ln_g, ln_b,
                                                    wq_sw, qh, k_sw, v_sw);

  window_kv_kn<<<2048, 256, 0, stream>>>(k_sw, v_sw, qh, kv_sw, zb);

  piece_mix_kv_kn<<<2048, 256, 0, stream>>>(pA, kv_sw, kvp_sw);
  piece_mix_z_kn<<<3136, 256, 0, stream>>>(pw, zb, zpb);

  attn_apply_kn<<<2048, 256, 0, stream>>>(qh, kvp_sw, zpb, attn);

  lepe_add_kn<<<100352, 256, 0, stream>>>(v_sw, attn, lw, lb, yb);

  out_gemm_kn<<<dim3(784, 4), 256, 0, stream>>>(yb, wo_sw, bout, out);
}
